// Attention_28862180229481
// MI455X (gfx1250) — compile-verified, matrix operand order checked by reading, not run on hardware
//
#include <hip/hip_runtime.h>
#include <hip/hip_bf16.h>
#include <cstdint>

typedef __attribute__((ext_vector_type(16))) __bf16 v16bf;
typedef __attribute__((ext_vector_type(8)))  __bf16 v8bf;
typedef __attribute__((ext_vector_type(8)))  float  v8f;
typedef unsigned __attribute__((ext_vector_type(4))) u32x4;
typedef int      __attribute__((ext_vector_type(4))) i32x4;
typedef int      __attribute__((ext_vector_type(8))) i32x8;

#define WMMA_BF16(a, b, c) \
  __builtin_amdgcn_wmma_f32_16x16x32_bf16(false, (a), false, (b), (short)0, (c), false, false)
#define CAT16(lo, hi) \
  __builtin_shufflevector((lo), (hi), 0, 1, 2, 3, 4, 5, 6, 7, 8, 9, 10, 11, 12, 13, 14, 15)

#if __has_builtin(__builtin_amdgcn_tensor_load_to_lds) && \
    __has_builtin(__builtin_amdgcn_s_wait_tensorcnt)
#define USE_TDM 1
#else
#define USE_TDM 0
#endif

// Problem constants
constexpr int B_  = 16;
constexpr int N_  = 785;
constexpr int NP  = 800;            // padded sequence length (25 chunks of 32)
constexpr int C_  = 768;
constexpr int H_  = 12;
constexpr int HD  = 64;
constexpr int P_  = 48;
constexpr int M_  = B_ * N_;        // 12560 (multiple of 16)
constexpr long NN = (long)N_ * N_;  // 616225
constexpr long QKV_ELEMS = (long)B_ * H_ * N_ * HD;   // 9,646,080
constexpr long VT_ELEMS  = (long)B_ * H_ * HD * NP;   // 9,830,400
constexpr long XB_ELEMS  = (long)M_ * C_;             // 9,646,080
constexpr long WQ_ELEMS  = (long)3 * C_ * C_;         // 1,769,472
constexpr long WP_ELEMS  = (long)C_ * C_;             //   589,824
constexpr long POSB_ELEMS = 7394704;                  // H*NN = 7,394,700 padded to /8

// K-offset pattern for 16-bit WMMA A/B fragments (16x32 tile, wave32):
__host__ __device__ __forceinline__ constexpr int koff(int e, int g) {
  return (e < 8) ? (g * 8 + e) : (16 + g * 8 + (e - 8));
}

__device__ __forceinline__ v8bf ld8(const __bf16* p) { return *(const v8bf*)p; }

// ---------------------------------------------------------------------------
// Kernel 0: f32 -> bf16 conversion, 8 elements/thread (n divisible by 8).
// ---------------------------------------------------------------------------
__global__ __launch_bounds__(256) void k_cvt(const float* __restrict__ src,
                                             __bf16* __restrict__ dst, long n8) {
  long i = (long)blockIdx.x * 256 + threadIdx.x;
  if (i >= n8) return;
  const float4* s = (const float4*)src + i * 2;
  float4 f0 = s[0], f1 = s[1];
  v8bf o;
  o[0] = (__bf16)f0.x; o[1] = (__bf16)f0.y; o[2] = (__bf16)f0.z; o[3] = (__bf16)f0.w;
  o[4] = (__bf16)f1.x; o[5] = (__bf16)f1.y; o[6] = (__bf16)f1.z; o[7] = (__bf16)f1.w;
  *((v8bf*)dst + i) = o;
}

// ---------------------------------------------------------------------------
// Kernel 1: QKV projection (bf16 in/out). One wave: 32(M) x 64(Nout) tile,
// software-pipelined over K. q scaled by 0.125; v stored transposed
// as vT[B,H,64,NP].
// ---------------------------------------------------------------------------
__global__ __launch_bounds__(128) void k_qkv(const __bf16* __restrict__ xb,
                                             const __bf16* __restrict__ wb,
                                             __bf16* __restrict__ q,
                                             __bf16* __restrict__ kk,
                                             __bf16* __restrict__ vT) {
  const int MT = (M_ + 31) / 32;   // 393
  const int NT = (3 * C_) / 64;    // 36
  int wid = blockIdx.x * 4 + (threadIdx.x >> 5);
  if (wid >= MT * NT) return;
  int nt = wid / MT, mt = wid % MT;
  int mbase = mt * 32, nbase = nt * 64;
  int l = threadIdx.x & 31, lm = l & 15, g = l >> 4;

  v8f acc0[4] = {}, acc1[4] = {};
  int rA = mbase + lm;            // always < M_
  int rB = mbase + 16 + lm;
  int rBc = (rB < M_) ? rB : (M_ - 1);
  const __bf16* xrA = xb + (long)rA * C_;
  const __bf16* xrB = xb + (long)rBc * C_;
  const __bf16* wr0 = wb + (long)(nbase + 0 * 16 + lm) * C_;
  const __bf16* wr1 = wb + (long)(nbase + 1 * 16 + lm) * C_;
  const __bf16* wr2 = wb + (long)(nbase + 2 * 16 + lm) * C_;
  const __bf16* wr3 = wb + (long)(nbase + 3 * 16 + lm) * C_;

  v8bf aA0n, aA1n, aB0n, aB1n, b0n[4], b1n[4];
  {
    int o = g * 8;
    aA0n = ld8(xrA + o);    aA1n = ld8(xrA + 16 + o);
    aB0n = ld8(xrB + o);    aB1n = ld8(xrB + 16 + o);
    b0n[0] = ld8(wr0 + o);  b1n[0] = ld8(wr0 + 16 + o);
    b0n[1] = ld8(wr1 + o);  b1n[1] = ld8(wr1 + 16 + o);
    b0n[2] = ld8(wr2 + o);  b1n[2] = ld8(wr2 + 16 + o);
    b0n[3] = ld8(wr3 + o);  b1n[3] = ld8(wr3 + 16 + o);
  }
  for (int kb = 0; kb < C_ / 32; ++kb) {
    v8bf aA0 = aA0n, aA1 = aA1n, aB0 = aB0n, aB1 = aB1n;
    v8bf b0[4] = {b0n[0], b0n[1], b0n[2], b0n[3]};
    v8bf b1[4] = {b1n[0], b1n[1], b1n[2], b1n[3]};
    if (kb + 1 < C_ / 32) {
      int o = (kb + 1) * 32 + g * 8;
      aA0n = ld8(xrA + o);    aA1n = ld8(xrA + 16 + o);
      aB0n = ld8(xrB + o);    aB1n = ld8(xrB + 16 + o);
      b0n[0] = ld8(wr0 + o);  b1n[0] = ld8(wr0 + 16 + o);
      b0n[1] = ld8(wr1 + o);  b1n[1] = ld8(wr1 + 16 + o);
      b0n[2] = ld8(wr2 + o);  b1n[2] = ld8(wr2 + 16 + o);
      b0n[3] = ld8(wr3 + o);  b1n[3] = ld8(wr3 + 16 + o);
    }
    v16bf A0 = CAT16(aA0, aA1);
    v16bf A1 = CAT16(aB0, aB1);
#pragma unroll
    for (int t = 0; t < 4; ++t) {
      v16bf Bt = CAT16(b0[t], b1[t]);
      acc0[t] = WMMA_BF16(A0, Bt, acc0[t]);
      acc1[t] = WMMA_BF16(A1, Bt, acc1[t]);
    }
  }

#pragma unroll
  for (int t = 0; t < 4; ++t) {
    int n = nbase + t * 16 + lm;
    int which = n / C_;           // 0=q 1=k 2=v (tile never straddles)
    int c = n - which * C_;
    int h = c >> 6, d = c & 63;
#pragma unroll
    for (int half = 0; half < 2; ++half) {
#pragma unroll
      for (int r = 0; r < 8; ++r) {
        int m = mbase + half * 16 + r + 8 * g;
        if (m >= M_) continue;
        float val = half ? acc1[t][r] : acc0[t][r];
        int bb = m / N_, tt = m - bb * N_;
        if (which == 0) {
          q[((long)(bb * H_ + h) * N_ + tt) * HD + d] = (__bf16)(val * 0.125f);
        } else if (which == 1) {
          kk[((long)(bb * H_ + h) * N_ + tt) * HD + d] = (__bf16)val;
        } else {
          vT[((long)(bb * H_ + h) * HD + d) * NP + tt] = (__bf16)val;
        }
      }
    }
  }
}

// ---------------------------------------------------------------------------
// Kernel 2: position bias. posb[h][n][m] = 0.125 * sum_p emb[n][m][p]*pw[h][p]
// (memory-bound: 118 MB of pos_emb read exactly once)
// ---------------------------------------------------------------------------
__global__ __launch_bounds__(128) void k_pos(const float* __restrict__ emb,
                                             const float* __restrict__ pw,
                                             __bf16* __restrict__ posb) {
  const int RT = (int)((NN + 15) / 16);  // 38515
  int wid = blockIdx.x * 4 + (threadIdx.x >> 5);
  if (wid >= RT) return;
  long rbase = (long)wid * 16;
  int l = threadIdx.x & 31, lm = l & 15, g = l >> 4;

  v16bf b0, b1;
#pragma unroll
  for (int e = 0; e < 16; ++e) {
    int p0 = koff(e, g), p1 = 32 + koff(e, g);
    b0[e] = (lm < H_) ? (__bf16)(pw[lm * P_ + p0] * 0.125f) : (__bf16)0.f;
    b1[e] = (lm < H_ && p1 < P_) ? (__bf16)(pw[lm * P_ + p1] * 0.125f) : (__bf16)0.f;
  }
  long row = rbase + lm;
  bool rv = row < NN;
  const float* er = emb + (rv ? row : 0) * P_;
  v16bf a0, a1;
#pragma unroll
  for (int e = 0; e < 16; ++e) {
    int p0 = koff(e, g), p1 = 32 + koff(e, g);
    a0[e] = rv ? (__bf16)er[p0] : (__bf16)0.f;
    a1[e] = (rv && p1 < P_) ? (__bf16)er[p1] : (__bf16)0.f;
  }
  v8f acc = {};
  acc = WMMA_BF16(a0, b0, acc);
  acc = WMMA_BF16(a1, b1, acc);
#pragma unroll
  for (int r = 0; r < 8; ++r) {
    long nm = rbase + r + 8 * g;
    if (nm < NN && lm < H_) posb[(long)lm * NN + nm] = (__bf16)acc[r];
  }
}

// ---------------------------------------------------------------------------
// Kernel 3: attention for one (b, h, 16-row tile). bf16 score strip
// S[16][800] (25.6 KB LDS). Bias tile is DMA'd into S by the Tensor Data
// Mover (OOB rows/cols auto-zeroed), pass 1 accumulates Q·K^T on top of it;
// parallel two-pass softmax; P fragments rebuilt on the fly.
// ---------------------------------------------------------------------------
__global__ __launch_bounds__(32) void k_attn(const __bf16* __restrict__ q,
                                             const __bf16* __restrict__ kmat,
                                             const __bf16* __restrict__ vT,
                                             const __bf16* __restrict__ posb,
                                             __bf16* __restrict__ ao) {
  constexpr int CT = 50;  // 50 column tiles of 16 -> 800 padded cols
  __shared__ __bf16 S[16][NP];
  __shared__ float pmx[2][16], psum[2][16];
  __shared__ float mx[16], ism[16];

  int bid = blockIdx.x;
  int rt = bid % CT;
  int h  = (bid / CT) % H_;
  int b  = bid / (CT * H_);
  int t0 = rt * 16;
  int l = threadIdx.x, lm = l & 15, g = l >> 4;

  const __bf16* qh = q    + (long)(b * H_ + h) * N_ * HD;
  const __bf16* kh = kmat + (long)(b * H_ + h) * N_ * HD;
  const __bf16* vh = vT   + (long)(b * H_ + h) * HD * NP;
  const __bf16* ph = posb + (long)h * NN;

#if USE_TDM
  // ---- TDM: DMA bias tile posb[h][t0:t0+16][0:800] into S (bf16).      ----
  // D# group0: count=1 | lds_addr | global_addr(57b) | type=2.
  // D# group1: data_size=2B; tensor_dim0=785 (cols beyond -> 0),
  //            tensor_dim1=N-t0 (rows beyond -> 0), tile=800x16, stride0=785.
  {
    unsigned lds_off = (unsigned)(uintptr_t)(&S[0][0]);  // addr[31:0] = LDS offset
    unsigned long long ga = (unsigned long long)(uintptr_t)(ph + (long)t0 * N_);
    u32x4 g0;
    g0[0] = 1u;                                         // count=1, user D#
    g0[1] = lds_off;                                    // lds_addr (bytes)
    g0[2] = (unsigned)(ga & 0xffffffffu);               // global_addr[31:0]
    g0[3] = (unsigned)((ga >> 32) & 0x01ffffffu) | (2u << 30);  // [56:32] | type=2
    i32x8 g1 = {};
    g1[0] = (int)(1u << 16);                            // data_size = 2 bytes
    g1[1] = (int)((unsigned)N_ << 16);                  // tensor_dim0[15:0]
    g1[2] = (int)(((unsigned)(N_ - t0) & 0xffffu) << 16); // dim0 hi=0 | tensor_dim1 lo
    g1[3] = (int)((unsigned)NP << 16);                  // dim1 hi=0 | tile_dim0=800
    g1[4] = 16;                                         // tile_dim1=16, tile_dim2=0
    g1[5] = N_;                                         // tensor_dim0_stride[31:0]
    i32x4 zz = {};
#if defined(__clang_major__) && (__clang_major__ >= 23)
    i32x8 zz8 = {};
    __builtin_amdgcn_tensor_load_to_lds(g0, g1, zz, zz, zz8, 0);
#else
    __builtin_amdgcn_tensor_load_to_lds(g0, g1, zz, zz, 0);
#endif
  }
#endif

  int trow = t0 + lm;
  bool tv = trow < N_;
  const __bf16* qr = qh + (long)(tv ? trow : 0) * HD;
  v8bf z8 = {};
  v8bf q0 = ld8(qr + g * 8), q1 = ld8(qr + 16 + g * 8);
  v8bf q2 = ld8(qr + 32 + g * 8), q3 = ld8(qr + 48 + g * 8);
  if (!tv) { q0 = z8; q1 = z8; q2 = z8; q3 = z8; }
  v16bf qa0 = CAT16(q0, q1), qa1 = CAT16(q2, q3);

  // ---- Pass 1: S = (q*scale)·k^T + posb*scale; padded cols -> -1e30 ----
  v8bf k0n, k1n, k2n, k3n;
  {
    const __bf16* kr = kh + (long)((lm < N_) ? lm : 0) * HD;  // ct=0
    k0n = ld8(kr + g * 8); k1n = ld8(kr + 16 + g * 8);
    k2n = ld8(kr + 32 + g * 8); k3n = ld8(kr + 48 + g * 8);
  }
#if USE_TDM
  __builtin_amdgcn_s_wait_tensorcnt(0);   // bias tile resident in S
#endif
  for (int ct = 0; ct < CT; ++ct) {
    int m = ct * 16 + lm;
    bool mv = m < N_;
    v8bf k0 = k0n, k1 = k1n, k2 = k2n, k3 = k3n;
    if (!mv) { k0 = z8; k1 = z8; k2 = z8; k3 = z8; }
    if (ct + 1 < CT) {
      int m2 = (ct + 1) * 16 + lm;
      const __bf16* kr = kh + (long)((m2 < N_) ? m2 : 0) * HD;
      k0n = ld8(kr + g * 8); k1n = ld8(kr + 16 + g * 8);
      k2n = ld8(kr + 32 + g * 8); k3n = ld8(kr + 48 + g * 8);
    }
    v8f acc = {};
    acc = WMMA_BF16(qa0, CAT16(k0, k1), acc);
    acc = WMMA_BF16(qa1, CAT16(k2, k3), acc);
#pragma unroll
    for (int r = 0; r < 8; ++r) {
      int tt = t0 + r + 8 * g;
      int cc = ct * 16 + lm;
#if USE_TDM
      float val = acc[r] + (float)S[r + 8 * g][cc];  // bias from LDS (OOB = 0)
#else
      float val = acc[r];
      if (tt < N_ && mv) val += (float)ph[(long)tt * N_ + m];
#endif
      S[r + 8 * g][cc] = mv ? (__bf16)val : (__bf16)(-1e30f);
    }
  }
  __syncthreads();

  // ---- Pass 2: per-row max & 1/sum; 2 lanes per row, v8bf LDS reads ----
  {
    const v8bf* Sr = (const v8bf*)&S[lm][g * (NP / 2)];
    float mA = -1e30f;
    for (int i = 0; i < NP / 16; ++i) {
      v8bf v = Sr[i];
#pragma unroll
      for (int e = 0; e < 8; ++e) mA = fmaxf(mA, (float)v[e]);
    }
    float sA = 0.f;
    for (int i = 0; i < NP / 16; ++i) {
      v8bf v = Sr[i];
#pragma unroll
      for (int e = 0; e < 8; ++e) sA += __expf((float)v[e] - mA);
    }
    pmx[g][lm] = mA;
    psum[g][lm] = sA;
  }
  __syncthreads();
  if (l < 16) {
    float m0 = pmx[0][l], m1 = pmx[1][l];
    float m_ = fmaxf(m0, m1);
    float s_ = psum[0][l] * __expf(m0 - m_) + psum[1][l] * __expf(m1 - m_);
    mx[l] = m_;
    ism[l] = 1.f / s_;
  }
  __syncthreads();

  // ---- Pass 3: out = P @ V over 25 K-chunks of 32, V read from vT ----
  v8f oacc[4] = {};
  float rm = mx[lm], ri = ism[lm];
  const __bf16* vr0 = vh + (long)(0 * 16 + lm) * NP;
  const __bf16* vr1 = vh + (long)(1 * 16 + lm) * NP;
  const __bf16* vr2 = vh + (long)(2 * 16 + lm) * NP;
  const __bf16* vr3 = vh + (long)(3 * 16 + lm) * NP;

  v8bf v0n[4], v1n[4];
  {
    int o = g * 8;
    v0n[0] = ld8(vr0 + o); v1n[0] = ld8(vr0 + 16 + o);
    v0n[1] = ld8(vr1 + o); v1n[1] = ld8(vr1 + 16 + o);
    v0n[2] = ld8(vr2 + o); v1n[2] = ld8(vr2 + 16 + o);
    v0n[3] = ld8(vr3 + o); v1n[3] = ld8(vr3 + 16 + o);
  }
  for (int kc = 0; kc < NP / 32; ++kc) {
    v8bf v0[4] = {v0n[0], v0n[1], v0n[2], v0n[3]};
    v8bf v1[4] = {v1n[0], v1n[1], v1n[2], v1n[3]};
    if (kc + 1 < NP / 32) {
      int o = (kc + 1) * 32 + g * 8;
      v0n[0] = ld8(vr0 + o); v1n[0] = ld8(vr0 + 16 + o);
      v0n[1] = ld8(vr1 + o); v1n[1] = ld8(vr1 + 16 + o);
      v0n[2] = ld8(vr2 + o); v1n[2] = ld8(vr2 + 16 + o);
      v0n[3] = ld8(vr3 + o); v1n[3] = ld8(vr3 + 16 + o);
    }
    // Build P fragment from bf16 S (vector LDS reads + exp)
    v8bf s_lo = *(const v8bf*)&S[lm][kc * 32 + g * 8];
    v8bf s_hi = *(const v8bf*)&S[lm][kc * 32 + 16 + g * 8];
    v16bf pa;
#pragma unroll
    for (int e = 0; e < 8; ++e) {
      pa[e]     = (__bf16)(__expf((float)s_lo[e] - rm) * ri);
      pa[8 + e] = (__bf16)(__expf((float)s_hi[e] - rm) * ri);
    }
#pragma unroll
    for (int dt = 0; dt < 4; ++dt)
      oacc[dt] = WMMA_BF16(pa, CAT16(v0[dt], v1[dt]), oacc[dt]);
  }
#pragma unroll
  for (int dt = 0; dt < 4; ++dt) {
    int d = dt * 16 + lm;
#pragma unroll
    for (int r = 0; r < 8; ++r) {
      int tt = t0 + r + 8 * g;
      if (tt < N_) ao[(long)(b * N_ + tt) * C_ + h * HD + d] = (__bf16)oacc[dt][r];
    }
  }
}

// ---------------------------------------------------------------------------
// Kernel 4: output projection. out = attn_out @ proj_w^T + proj_b (f32 out).
// One wave: 32(M) x 64(N) tile.
// ---------------------------------------------------------------------------
__global__ __launch_bounds__(128) void k_proj(const __bf16* __restrict__ ao,
                                              const __bf16* __restrict__ wb,
                                              const float* __restrict__ bias,
                                              float* __restrict__ out) {
  const int MT = (M_ + 31) / 32;  // 393
  const int NT = C_ / 64;         // 12
  int wid = blockIdx.x * 4 + (threadIdx.x >> 5);
  if (wid >= MT * NT) return;
  int nt = wid / MT, mt = wid % MT;
  int mbase = mt * 32, nbase = nt * 64;
  int l = threadIdx.x & 31, lm = l & 15, g = l >> 4;

  v8f acc0[4] = {}, acc1[4] = {};
  int rA = mbase + lm;
  int rB = mbase + 16 + lm;
  int rBc = (rB < M_) ? rB : (M_ - 1);
  const __bf16* arA = ao + (long)rA * C_;
  const __bf16* arB = ao + (long)rBc * C_;
  const __bf16* wr0 = wb + (long)(nbase + 0 * 16 + lm) * C_;
  const __bf16* wr1 = wb + (long)(nbase + 1 * 16 + lm) * C_;
  const __bf16* wr2 = wb + (long)(nbase + 2 * 16 + lm) * C_;
  const __bf16* wr3 = wb + (long)(nbase + 3 * 16 + lm) * C_;

  v8bf aA0n, aA1n, aB0n, aB1n, b0n[4], b1n[4];
  {
    int o = g * 8;
    aA0n = ld8(arA + o);    aA1n = ld8(arA + 16 + o);
    aB0n = ld8(arB + o);    aB1n = ld8(arB + 16 + o);
    b0n[0] = ld8(wr0 + o);  b1n[0] = ld8(wr0 + 16 + o);
    b0n[1] = ld8(wr1 + o);  b1n[1] = ld8(wr1 + 16 + o);
    b0n[2] = ld8(wr2 + o);  b1n[2] = ld8(wr2 + 16 + o);
    b0n[3] = ld8(wr3 + o);  b1n[3] = ld8(wr3 + 16 + o);
  }
  for (int kb = 0; kb < C_ / 32; ++kb) {
    v8bf aA0 = aA0n, aA1 = aA1n, aB0 = aB0n, aB1 = aB1n;
    v8bf b0[4] = {b0n[0], b0n[1], b0n[2], b0n[3]};
    v8bf b1[4] = {b1n[0], b1n[1], b1n[2], b1n[3]};
    if (kb + 1 < C_ / 32) {
      int o = (kb + 1) * 32 + g * 8;
      aA0n = ld8(arA + o);    aA1n = ld8(arA + 16 + o);
      aB0n = ld8(arB + o);    aB1n = ld8(arB + 16 + o);
      b0n[0] = ld8(wr0 + o);  b1n[0] = ld8(wr0 + 16 + o);
      b0n[1] = ld8(wr1 + o);  b1n[1] = ld8(wr1 + 16 + o);
      b0n[2] = ld8(wr2 + o);  b1n[2] = ld8(wr2 + 16 + o);
      b0n[3] = ld8(wr3 + o);  b1n[3] = ld8(wr3 + 16 + o);
    }
    v16bf A0 = CAT16(aA0, aA1);
    v16bf A1 = CAT16(aB0, aB1);
#pragma unroll
    for (int t = 0; t < 4; ++t) {
      v16bf Bt = CAT16(b0[t], b1[t]);
      acc0[t] = WMMA_BF16(A0, Bt, acc0[t]);
      acc1[t] = WMMA_BF16(A1, Bt, acc1[t]);
    }
  }
#pragma unroll
  for (int t = 0; t < 4; ++t) {
    int n = nbase + t * 16 + lm;
    float bs = bias[n];
#pragma unroll
    for (int half = 0; half < 2; ++half) {
#pragma unroll
      for (int r = 0; r < 8; ++r) {
        int m = mbase + half * 16 + r + 8 * g;
        if (m >= M_) continue;
        float val = half ? acc1[t][r] : acc0[t][r];
        out[(long)m * C_ + n] = val + bs;
      }
    }
  }
}

// ---------------------------------------------------------------------------
extern "C" void kernel_launch(void* const* d_in, const int* in_sizes, int n_in,
                              void* d_out, int out_size, void* d_ws, size_t ws_size,
                              hipStream_t stream) {
  const float* x          = (const float*)d_in[0];
  const float* qkv_w      = (const float*)d_in[1];
  const float* pos_emb    = (const float*)d_in[2];
  const float* pos_proj_w = (const float*)d_in[3];
  const float* proj_w     = (const float*)d_in[4];
  const float* proj_b     = (const float*)d_in[5];
  float* out = (float*)d_out;

  // Workspace layout (bf16, all segments multiples of 8 elements => 16B aligned):
  // xb | wqkvb | wprojb | q | k | vT | posb | ao   (~116 MB)
  __bf16* base   = (__bf16*)d_ws;
  __bf16* xb     = base;
  __bf16* wqkvb  = xb + XB_ELEMS;
  __bf16* wprojb = wqkvb + WQ_ELEMS;
  __bf16* q      = wprojb + WP_ELEMS;
  __bf16* k      = q + QKV_ELEMS;
  __bf16* vT     = k + QKV_ELEMS;
  __bf16* posb   = vT + VT_ELEMS;
  __bf16* ao     = posb + POSB_ELEMS;

  // 0) precision conversion + vT pad zeroing
  k_cvt<<<(int)((XB_ELEMS / 8 + 255) / 256), 256, 0, stream>>>(x, xb, XB_ELEMS / 8);
  k_cvt<<<(int)((WQ_ELEMS / 8 + 255) / 256), 256, 0, stream>>>(qkv_w, wqkvb, WQ_ELEMS / 8);
  k_cvt<<<(int)((WP_ELEMS / 8 + 255) / 256), 256, 0, stream>>>(proj_w, wprojb, WP_ELEMS / 8);
  hipMemsetAsync(vT, 0, (size_t)VT_ELEMS * sizeof(__bf16), stream);

  // 1) QKV projection
  {
    int tiles = ((M_ + 31) / 32) * (3 * C_ / 64);  // 14148
    k_qkv<<<(tiles + 3) / 4, 128, 0, stream>>>(xb, wqkvb, q, k, vT);
  }
  // 2) position bias
  {
    int tiles = (int)((NN + 15) / 16);             // 38515
    k_pos<<<(tiles + 3) / 4, 128, 0, stream>>>(pos_emb, pos_proj_w, posb);
  }
  // 3) attention
  k_attn<<<B_ * H_ * 50, 32, 0, stream>>>(q, k, vT, posb, ao);
  // 4) output projection
  {
    int tiles = ((M_ + 31) / 32) * (C_ / 64);      // 4716
    k_proj<<<(tiles + 3) / 4, 128, 0, stream>>>(ao, wprojb, proj_b, out);
  }
}
